// NeedlemanWunschAligner_17222818857424
// MI455X (gfx1250) — compile-verified, matrix operand order checked by reading, not run on hardware
//
#include <hip/hip_runtime.h>
#include <hip/hip_bf16.h>

typedef __attribute__((ext_vector_type(16))) _Float16 v16h;
typedef __attribute__((ext_vector_type(8)))  float    v8f;

#define NN 512
#define ALPHA 22
#define NEGV (-1.0e8f)

// ---------------------------------------------------------------------------
// Kernel 1: P[22x512] = W_embed[22x512] @ W_proj[512x512] + b_proj  (WMMA f16)
// One wave per 16x16 tile; M padded 22->32. Branch-free fragment loads:
// clamped row index + value mask so A loads vectorize to b128 clauses.
// ---------------------------------------------------------------------------
__global__ __launch_bounds__(32)
void compute_P_wmma(const float* __restrict__ We, const float* __restrict__ Wp,
                    const float* __restrict__ bp, float* __restrict__ P) {
    const int lane = threadIdx.x;
    const int half = lane >> 4;
    const int l16  = lane & 15;
    const int mrow = blockIdx.y * 16 + l16;   // A row held by this lane
    const int ncol = blockIdx.x * 16 + l16;   // B col held by this lane
    const int arow  = (mrow < ALPHA) ? mrow : 0;
    const float amask = (mrow < ALPHA) ? 1.0f : 0.0f;
    const float* __restrict__ Arow = We + arow * NN;

    v8f acc = {};
    for (int k0 = 0; k0 < NN; k0 += 32) {
        const int kb = k0 + half * 8;    // K groups: {kb..kb+7} and {kb+16..kb+23}
        float a0[8], a1[8], b0[8], b1[8];
#pragma unroll
        for (int q = 0; q < 8; ++q) {    // contiguous -> global_load_b128 pairs
            a0[q] = Arow[kb + q];
            a1[q] = Arow[kb + 16 + q];
        }
#pragma unroll
        for (int q = 0; q < 8; ++q) {    // strided column reads (b32 clause)
            b0[q] = Wp[(kb + q) * NN + ncol];
            b1[q] = Wp[(kb + 16 + q) * NN + ncol];
        }
        v16h a, b;
#pragma unroll
        for (int q = 0; q < 8; ++q) {
            a[q]     = (_Float16)(a0[q] * amask);
            a[q + 8] = (_Float16)(a1[q] * amask);
            b[q]     = (_Float16)b0[q];
            b[q + 8] = (_Float16)b1[q];
        }
        acc = __builtin_amdgcn_wmma_f32_16x16x32_f16(
            false, a, false, b, (short)0, acc, false, false);
    }

    const float bias = bp[ncol];
#pragma unroll
    for (int r = 0; r < 8; ++r) {
        const int m = blockIdx.y * 16 + half * 8 + r;
        if (m < ALPHA) P[m * NN + ncol] = acc[r] + bias;
    }
}

// ---------------------------------------------------------------------------
// Kernel 2: G = P P^T (22x22), qx[a] = P[a]·Wg[0:512], qy[a] = P[a]·Wg[512:]
// ---------------------------------------------------------------------------
__global__ __launch_bounds__(512)
void compute_Gq(const float* __restrict__ P, const float* __restrict__ Wg,
                float* __restrict__ G, float* __restrict__ qx, float* __restrict__ qy) {
    const int t = threadIdx.x;
    if (t < ALPHA * ALPHA) {
        const int i = t / ALPHA, j = t % ALPHA;
        float s = 0.f;
        for (int k = 0; k < NN; ++k) s += P[i * NN + k] * P[j * NN + k];
        G[t] = s;
    }
    if (t < ALPHA) {
        float s = 0.f;
        for (int k = 0; k < NN; ++k) s += P[t * NN + k] * Wg[k];
        qx[t] = s;
    }
    if (t >= 64 && t < 64 + ALPHA) {
        const int a = t - 64;
        float s = 0.f;
        for (int k = 0; k < NN; ++k) s += P[a * NN + k] * Wg[NN + k];
        qy[a] = s;
    }
}

// ---------------------------------------------------------------------------
// Kernel 3: soft NW forward + backward posterior. One workgroup per batch,
// 512 threads; thread t owns row i=t+1 for the whole sweep (loop-invariant
// G-row bases). Forward: LDS diagonal ping-pong, V spilled to L2 workspace.
// Backward: V diagonals also ping-ponged in LDS; one prefetched global load
// per cell (own V), successor V/E all from LDS.
// ---------------------------------------------------------------------------
__global__ __launch_bounds__(512)
void nw_grad(const int* __restrict__ x, const int* __restrict__ y,
             const float* __restrict__ G, const float* __restrict__ qx,
             const float* __restrict__ qy, const float* __restrict__ bg,
             float* __restrict__ V, float* __restrict__ out) {
    __shared__ int   xs[NN], ys[NN];
    __shared__ float Gs[ALPHA * ALPHA];
    __shared__ float red[NN];
    __shared__ float bufE[3][516];
    __shared__ float bufV[3][516];

    const int b = blockIdx.x;
    const int t = threadIdx.x;
    const int i = t + 1;                               // owned row, 1..512
    float* __restrict__ Vb = V + (size_t)b * NN * NN;

    // ---- stage indices + score table; gap score A via tree reduction ----
    const int xt = x[b * NN + t];
    const int yt = y[b * NN + t];
    xs[t] = xt;  ys[t] = yt;
    if (t < ALPHA * ALPHA) Gs[t] = G[t];
    red[t] = qx[xt] + qy[yt];
    __syncthreads();
    for (int s = 256; s > 0; s >>= 1) {
        if (t < s) red[t] += red[t + s];
        __syncthreads();
    }
    const float A = red[0] * (1.0f / (float)NN) + bg[0];

    // loop-invariant per-thread lookups
    const int xi   = xs[t];                            // xs[i-1]
    const int xip1 = (i < NN) ? xs[i] : 0;             // xs[i] (successor rows)
    const float* __restrict__ Gi   = Gs + xi   * ALPHA; // theta(i, *)
    const float* __restrict__ Gip1 = Gs + xip1 * ALPHA; // theta(i+1, *)

    // ---- forward: V[i,j] = theta + logsumexp(diag, up+A, left+A) ----
    for (int s = t; s < 516; s += NN) {
        bufE[0][s] = (s == 0) ? 0.0f : NEGV;   // diagonal d=0
        bufE[1][s] = NEGV;                     // diagonal d=1
        bufE[2][s] = NEGV;
    }
    __syncthreads();

    int p2 = 0, p1 = 1, c = 2;
    for (int d = 2; d <= 2 * NN; ++d) {
        const int j = d - i;
        float v = NEGV;
        if (j >= 1 && j <= NN) {
            const float diag = bufE[p2][t];            // V[i-1,j-1]
            const float up   = bufE[p1][t]     + A;    // V[i-1,j] + A
            const float left = bufE[p1][t + 1] + A;    // V[i,j-1] + A
            const float m = fmaxf(diag, fmaxf(up, left));
            const float lse = m + __logf(__expf(diag - m) + __expf(up - m) + __expf(left - m));
            v = Gi[ys[j - 1]] + lse;
            Vb[(t << 9) + (j - 1)] = v;
        }
        bufE[c][i] = v;
        if (t == 0) bufE[c][0] = NEGV;                 // border V[0,d]
        __syncthreads();
        const int tmp = p2; p2 = p1; p1 = c; c = tmp;
    }

    __threadfence_block();   // make Vb stores visible within the workgroup
    __syncthreads();

    // ---- backward: E[i,j] = sum of successor E * softmax weight (<= 1) ----
    for (int s = t; s < 516; s += NN) {
        bufE[0][s] = 0.0f;  bufE[1][s] = 0.0f;  bufE[2][s] = 0.0f;
        bufV[0][s] = 0.0f;  bufV[1][s] = 0.0f;  bufV[2][s] = 0.0f;
    }
    __syncthreads();

    // prefetch own V for first diagonal (d = 1024)
    float v_next = 0.0f;
    {
        const int j0 = 2 * NN - i;
        if (j0 >= 1 && j0 <= NN) v_next = Vb[(t << 9) + (j0 - 1)];
    }

    p2 = 0; p1 = 1; c = 2;
    for (int d = 2 * NN; d >= 2; --d) {
        const int j = d - i;
        const float vij = v_next;
        // prefetch next diagonal's own V (latency hidden behind this step)
        {
            const int jn = j - 1;
            if (jn >= 1 && jn <= NN) v_next = Vb[(t << 9) + (jn - 1)];
        }
        float e = 0.0f;
        float vstore = 0.0f;
        if (j >= 1 && j <= NN) {
            vstore = vij;
            if (i == NN && j == NN) {
                e = 1.0f;
            } else {
                const int yjm1 = ys[j - 1];
                const int yj   = (j < NN) ? ys[j] : 0;
                if (i < NN && j < NN)   // diag successor (i+1, j+1)
                    e += bufE[p2][t + 2] * __expf(vij - bufV[p2][t + 2] + Gip1[yj]);
                if (i < NN)             // up successor (i+1, j)
                    e += bufE[p1][t + 2] * __expf(vij + A - bufV[p1][t + 2] + Gip1[yjm1]);
                if (j < NN)             // left successor (i, j+1)
                    e += bufE[p1][t + 1] * __expf(vij + A - bufV[p1][t + 1] + Gi[yj]);
            }
            out[((size_t)b << 18) + (t << 9) + (j - 1)] = e;
        }
        bufE[c][i] = e;
        bufV[c][i] = vstore;
        __syncthreads();
        const int tmp = p2; p2 = p1; p1 = c; c = tmp;
    }
}

// ---------------------------------------------------------------------------
extern "C" void kernel_launch(void* const* d_in, const int* in_sizes, int n_in,
                              void* d_out, int out_size, void* d_ws, size_t ws_size,
                              hipStream_t stream) {
    const int*   x  = (const int*)d_in[0];      // [16,512]
    const int*   y  = (const int*)d_in[1];      // [16,512]
    const float* We = (const float*)d_in[2];    // [22,512]
    const float* Wp = (const float*)d_in[3];    // [512,512]
    const float* bp = (const float*)d_in[4];    // [512]
    const float* Wg = (const float*)d_in[5];    // [1024]
    const float* bg = (const float*)d_in[6];    // [1]
    float* out = (float*)d_out;                 // [16,512,512]

    float* ws = (float*)d_ws;
    float* P  = ws;                 // 22*512 floats
    float* G  = ws + 12288;         // 484 floats
    float* qx = ws + 13056;         // 22 floats
    float* qy = ws + 13184;         // 22 floats
    float* V  = ws + 16384;         // 16*512*512 floats (~16.8 MB, L2-resident)

    compute_P_wmma<<<dim3(32, 2), 32, 0, stream>>>(We, Wp, bp, P);
    compute_Gq<<<1, 512, 0, stream>>>(P, Wg, G, qx, qy);
    nw_grad<<<16, 512, 0, stream>>>(x, y, G, qx, qy, bg, V, out);
}
